// sparseFeaturePnP_72370198937666
// MI455X (gfx1250) — matrix operand order repaired and checked by reading, not compiled, not run on hardware
//
#include <hip/hip_runtime.h>
#include <math.h>

// ---------------- problem constants (match reference) ----------------
#define CC   128
#define HH   480
#define WW   480
#define N_ITERS 10
#define LAMBDA_ 0.01f
#define LR_     1.0f

typedef _Float16 v16h __attribute__((ext_vector_type(16)));
typedef float    v8f  __attribute__((ext_vector_type(8)));
typedef float    v4f  __attribute__((ext_vector_type(4)));

static constexpr int HW = HH * WW;

// Extract element i (runtime) from a v8f accumulator via cndmask chain.
__device__ __forceinline__ float vpick(v8f v, int i) {
    float r = v[0];
#pragma unroll
    for (int j = 1; j < 8; ++j) r = (i == j) ? (float)v[j] : r;
    return r;
}

// Convert the canonical per-lane layout (contiguous k: lane L holds column
// (L&15), K = (L<16 ? 0..15 : 16..31) packed sequentially) into the A-matrix
// 16-bit layout (lane<16 holds K 0..7 & 16..23; lane>=16 holds K 8..15 & 24..31)
// by swapping half of the data with lane^16.  (CDNA5 ISA 7.12.2)
__device__ __forceinline__ v16h make_a_layout(v16h can, int lane) {
    union U { v16h h; int d[8]; };
    U own; own.h = can;
    U pr;
#pragma unroll
    for (int i = 0; i < 8; ++i) pr.d[i] = __shfl_xor(own.d[i], 16, 32);
    U res;
    if (lane < 16) {
#pragma unroll
        for (int i = 0; i < 4; ++i) { res.d[i] = own.d[i]; res.d[i + 4] = pr.d[i]; }
    } else {
#pragma unroll
        for (int i = 0; i < 4; ++i) { res.d[i] = pr.d[i + 4]; res.d[i + 4] = own.d[i + 4]; }
    }
    return res.h;
}

// ---------------- init: pose <- (R_init, t_init) ----------------
__global__ void pnp_init_pose(const float* __restrict__ Rinit,
                              const float* __restrict__ tinit,
                              float* __restrict__ pose) {
    int t = threadIdx.x;
    if (t < 9)        pose[t] = Rinit[t];
    else if (t < 12)  pose[t] = tinit[t - 9];
}

// ---------------- one-time layout transform: (C,H,W) -> [y][x][map][c] ------
// Channel-contiguous layout: every bilinear corner read becomes a fully used
// 64B line instead of 4 useful bytes out of a line per channel.
__global__ __launch_bounds__(256)
void pnp_transpose(const float* __restrict__ src, float* __restrict__ dst, int map) {
    __shared__ float tile[32][33];
    int b  = blockIdx.x;
    int y  = b % HH;            b /= HH;
    int x0 = (b % (WW / 32)) * 32; b /= (WW / 32);
    int c0 = b * 32;
    const int t = threadIdx.x;
#pragma unroll
    for (int i = 0; i < 4; ++i) {           // load: lanes over x (coalesced)
        int idx = t + i * 256;
        int cl = idx >> 5, xl = idx & 31;
        tile[cl][xl] = src[(size_t)(c0 + cl) * HW + (size_t)y * WW + x0 + xl];
    }
    __syncthreads();
#pragma unroll
    for (int i = 0; i < 4; ++i) {           // store: lanes over c (coalesced)
        int idx = t + i * 256;
        int xl = idx >> 5, cl = idx & 31;
        dst[((size_t)y * WW + x0 + xl) * (3 * CC) + (size_t)map * CC + c0 + cl] = tile[cl][xl];
    }
}

// ---------------- per-iteration gather + WMMA Gram + partial reduce ----------
// Block = 256 threads = 8 waves; wave w handles point tile (blockIdx*8 + w),
// each tile = 16 points.  Lane L owns point (L&15) and channel half
// (L<16 ? low : high) of each 32-channel K-chunk.
template <bool TRANS>
__global__ __launch_bounds__(256)
void pnp_accum(const float* __restrict__ pts3D,
               const float* __restrict__ fref,
               const float* __restrict__ fq,
               const float* __restrict__ fgx,
               const float* __restrict__ fgy,
               const float* __restrict__ T,
               const float* __restrict__ Km,
               const float* __restrict__ pose,
               float* __restrict__ partials,
               const int* __restrict__ imwP,
               const int* __restrict__ imhP,
               int nPts) {
    __shared__ float diagLds[8][5][16];
    __shared__ float wp[8][27];

    const int tid  = threadIdx.x;
    const int wave = tid >> 5;
    const int lane = tid & 31;
    const int p    = lane & 15;
    const bool hiK = (lane >= 16);

    const int tile = blockIdx.x * 8 + wave;
    const int n0   = tile * 16;
    const bool active = (n0 < nPts);

    const float R00 = pose[0], R01 = pose[1], R02 = pose[2];
    const float R10 = pose[3], R11 = pose[4], R12 = pose[5];
    const float R20 = pose[6], R21 = pose[7], R22 = pose[8];
    const float T0 = pose[9], T1 = pose[10], T2 = pose[11];
    const float fx = Km[0], fy = Km[4], cx = Km[2], cy = Km[5];
    const float imwF = (float)(*imwP);
    const float imhF = (float)(*imhP);

    float X = 0.f, Y = 0.f, Z = 0.f, zs = 1e-6f, m = 0.f;
    v8f cEGx = {}, cEGy = {}, cXX = {}, cXY = {}, cYY = {};

    if (active) {
        const int nidx  = n0 + p;
        const int nsafe = (nidx < nPts) ? nidx : (nPts - 1);
        const bool realPt = (nidx < nPts);

        const float px = pts3D[nsafe * 3 + 0];
        const float py = pts3D[nsafe * 3 + 1];
        const float pz = pts3D[nsafe * 3 + 2];
        X = R00 * px + R01 * py + R02 * pz + T0;
        Y = R10 * px + R11 * py + R12 * pz + T1;
        Z = R20 * px + R21 * py + R22 * pz + T2;
        zs = (Z > 1e-6f) ? Z : 1e-6f;
        const float u = fx * X / zs + cx;
        const float v = fy * Y / zs + cy;
        m = (realPt && u >= 0.f && v >= 0.f && u < imwF && v < imhF && Z > 1e-6f) ? 1.f : 0.f;

        const float uc = fminf(fmaxf(u, 0.f), (float)(WW - 1));
        const float vc = fminf(fmaxf(v, 0.f), (float)(HH - 1));
        const float x0 = fminf(fmaxf(floorf(uc), 0.f), (float)(WW - 2));
        const float y0 = fminf(fmaxf(floorf(vc), 0.f), (float)(HH - 2));
        const int xi = (int)x0, yi = (int)y0;
        const float dx = uc - x0, dy = vc - y0;
        const float w00 = (1.f - dy) * (1.f - dx);
        const float w01 = (1.f - dy) * dx;
        const float w10 = dy * (1.f - dx);
        const float w11 = dy * dx;
        const int baseIdx = yi * WW + xi;

        const float* refRow = fref + (size_t)nsafe * CC;

#pragma unroll 1
        for (int kb = 0; kb < CC; kb += 32) {
            const int k0 = kb + (hiK ? 16 : 0);
            v16h eC, gxC, gyC;   // canonical contiguous layout: element e = channel k0 + e
            if constexpr (TRANS) {
                // single base address; all corner/map offsets are immediates
                const float* bT = T + (size_t)baseIdx * (3 * CC) + k0;
#pragma unroll
                for (int g = 0; g < 4; ++g) {
                    auto ld4 = [&](int mp, int dyy, int dxx) -> v4f {
                        return *reinterpret_cast<const v4f*>(
                            bT + ((size_t)dyy * WW * 3 + (size_t)dxx * 3 + mp) * CC + g * 4);
                    };
                    const v4f q00 = ld4(0, 0, 0), q01 = ld4(0, 0, 1), q10 = ld4(0, 1, 0), q11 = ld4(0, 1, 1);
                    const v4f a00 = ld4(1, 0, 0), a01 = ld4(1, 0, 1), a10 = ld4(1, 1, 0), a11 = ld4(1, 1, 1);
                    const v4f b00 = ld4(2, 0, 0), b01 = ld4(2, 0, 1), b10 = ld4(2, 1, 0), b11 = ld4(2, 1, 1);
                    const v4f fv  = w00 * q00 + w01 * q01 + w10 * q10 + w11 * q11;
                    const v4f gxv = w00 * a00 + w01 * a01 + w10 * a10 + w11 * a11;
                    const v4f gyv = w00 * b00 + w01 * b01 + w10 * b10 + w11 * b11;
                    const v4f rf  = *reinterpret_cast<const v4f*>(refRow + k0 + g * 4);
                    const v4f ev  = fv - rf;
#pragma unroll
                    for (int j = 0; j < 4; ++j) {
                        eC[g * 4 + j]  = (_Float16)ev[j];
                        gxC[g * 4 + j] = (_Float16)gxv[j];
                        gyC[g * 4 + j] = (_Float16)gyv[j];
                    }
                }
            } else {
#pragma unroll
                for (int e = 0; e < 16; ++e) {
                    const size_t co = (size_t)(k0 + e) * HW + baseIdx;
                    const float* q0  = fq  + co;
                    const float* x0p = fgx + co;
                    const float* y0p = fgy + co;
                    float fv = w00 * q0[0]  + w01 * q0[1]  + w10 * q0[WW]  + w11 * q0[WW + 1];
                    float gx = w00 * x0p[0] + w01 * x0p[1] + w10 * x0p[WW] + w11 * x0p[WW + 1];
                    float gy = w00 * y0p[0] + w01 * y0p[1] + w10 * y0p[WW] + w11 * y0p[WW + 1];
                    float ev = fv - refRow[k0 + e];
                    eC[e]  = (_Float16)ev;
                    gxC[e] = (_Float16)gx;
                    gyC[e] = (_Float16)gy;
                }
            }
            // Only Gx/Gy are used as A, in the same contiguous k order as B: diag(E.Gx^T) == diag(Gx.E^T).
            const v16h AGx = gxC;
            const v16h AGy = gyC;
            cEGx = __builtin_amdgcn_wmma_f32_16x16x32_f16(false, AGx, false, eC,  (short)0, cEGx, false, false);
            cEGy = __builtin_amdgcn_wmma_f32_16x16x32_f16(false, AGy, false, eC,  (short)0, cEGy, false, false);
            cXX  = __builtin_amdgcn_wmma_f32_16x16x32_f16(false, AGx, false, gxC, (short)0, cXX,  false, false);
            cXY  = __builtin_amdgcn_wmma_f32_16x16x32_f16(false, AGx, false, gyC, (short)0, cXY,  false, false);
            cYY  = __builtin_amdgcn_wmma_f32_16x16x32_f16(false, AGy, false, gyC, (short)0, cYY,  false, false);
        }

        // diag(n,n): lane<8 -> point=lane, vgpr=lane; lane>=24 -> point=lane-16, vgpr=lane-24
        const bool hold = (lane < 8) || (lane >= 24);
        if (hold) {
            const int dl = (lane < 8) ? lane : (lane - 24);
            const int pt = (lane < 8) ? lane : (lane - 16);
            diagLds[wave][0][pt] = vpick(cEGx, dl);
            diagLds[wave][1][pt] = vpick(cEGy, dl);
            diagLds[wave][2][pt] = vpick(cXX,  dl);
            diagLds[wave][3][pt] = vpick(cXY,  dl);
            diagLds[wave][4][pt] = vpick(cYY,  dl);
        }
    }
    __syncthreads();

    float vals[27];
#pragma unroll
    for (int i = 0; i < 27; ++i) vals[i] = 0.f;

    if (active && lane < 16) {
        const float segx = diagLds[wave][0][p] * m;
        const float segy = diagLds[wave][1][p] * m;
        const float sxx  = diagLds[wave][2][p] * m;
        const float sxy  = diagLds[wave][3][p] * m;
        const float syy  = diagLds[wave][4][p] * m;
        const float iz = 1.f / zs;
        const float a0 = fx * iz, a2 = -fx * X * iz * iz;
        const float b1 = fy * iz, b2 = -fy * Y * iz * iz;
        const float u6[6] = { a2 * Y, a0 * Z - a2 * X, -a0 * Y, a0, 0.f, a2 };
        const float w6[6] = { b2 * Y - b1 * Z, -b2 * X, b1 * X, 0.f, b1, b2 };
#pragma unroll
        for (int k = 0; k < 6; ++k) vals[k] = u6[k] * segx + w6[k] * segy;
        int idx = 6;
#pragma unroll
        for (int k = 0; k < 6; ++k) {
#pragma unroll
            for (int l = k; l < 6; ++l) {
                vals[idx++] = sxx * u6[k] * u6[l]
                            + sxy * (u6[k] * w6[l] + w6[k] * u6[l])
                            + syy * w6[k] * w6[l];
            }
        }
    }

    // deterministic wave reduction (lanes >=16 carry zeros)
#pragma unroll
    for (int i = 0; i < 27; ++i) {
        float s = vals[i];
        s += __shfl_down(s, 16, 32);
        s += __shfl_down(s, 8, 32);
        s += __shfl_down(s, 4, 32);
        s += __shfl_down(s, 2, 32);
        s += __shfl_down(s, 1, 32);
        if (lane == 0) wp[wave][i] = s;
    }
    __syncthreads();

    if (tid < 27) {
        float s = 0.f;
#pragma unroll
        for (int w = 0; w < 8; ++w) s += wp[w][tid];
        partials[(size_t)blockIdx.x * 27 + tid] = s;
    }
}

// ---------------- reduce partials, solve 6x6, SO(3)-exp pose update ----------
__global__ void pnp_solve_update(const float* __restrict__ partials, int nBlocks,
                                 float* __restrict__ pose, float* __restrict__ outp) {
    __shared__ float acc[27];
    const int t = threadIdx.x;
    if (t < 27) {
        float s = 0.f;
        for (int b = 0; b < nBlocks; ++b) s += partials[(size_t)b * 27 + t];
        acc[t] = s;
    }
    __syncthreads();
    if (t == 0) {
        float g[6];
#pragma unroll
        for (int k = 0; k < 6; ++k) g[k] = acc[k];
        float Hm[6][6];
        int idx = 6;
        for (int k = 0; k < 6; ++k)
            for (int l = k; l < 6; ++l) { float hv = acc[idx++]; Hm[k][l] = hv; Hm[l][k] = hv; }
        for (int k = 0; k < 6; ++k) Hm[k][k] += (Hm[k][k] + 1e-9f) * LAMBDA_;

        float A[6][7];
        for (int r = 0; r < 6; ++r) { for (int c = 0; c < 6; ++c) A[r][c] = Hm[r][c]; A[r][6] = g[r]; }
        for (int c = 0; c < 6; ++c) {
            int piv = c; float pv = fabsf(A[c][c]);
            for (int r = c + 1; r < 6; ++r) { float av = fabsf(A[r][c]); if (av > pv) { pv = av; piv = r; } }
            if (piv != c)
                for (int cc = c; cc < 7; ++cc) { float tmp = A[c][cc]; A[c][cc] = A[piv][cc]; A[piv][cc] = tmp; }
            const float inv = 1.f / A[c][c];
            for (int r = c + 1; r < 6; ++r) {
                const float f = A[r][c] * inv;
                for (int cc = c; cc < 7; ++cc) A[r][cc] -= f * A[c][cc];
            }
        }
        float x[6];
        for (int r = 5; r >= 0; --r) {
            float s = A[r][6];
            for (int cc = r + 1; cc < 6; ++cc) s -= A[r][cc] * x[cc];
            x[r] = s / A[r][r];
        }
        float delta[6];
#pragma unroll
        for (int k = 0; k < 6; ++k) delta[k] = -LR_ * x[k];

        const float wx = delta[0], wy = delta[1], wz = delta[2];
        const float th2 = wx * wx + wy * wy + wz * wz;
        const float th  = sqrtf(th2 + 1e-12f);
        const float Aa  = sinf(th) / th;
        const float Bb  = (1.f - cosf(th)) / (th2 + 1e-12f);
        const float Wm[3][3] = { {0.f, -wz, wy}, {wz, 0.f, -wx}, {-wy, wx, 0.f} };
        float W2[3][3];
        for (int i = 0; i < 3; ++i)
            for (int j = 0; j < 3; ++j) {
                float s = 0.f;
                for (int k = 0; k < 3; ++k) s += Wm[i][k] * Wm[k][j];
                W2[i][j] = s;
            }
        float Rd[3][3];
        for (int i = 0; i < 3; ++i)
            for (int j = 0; j < 3; ++j)
                Rd[i][j] = ((i == j) ? 1.f : 0.f) + Aa * Wm[i][j] + Bb * W2[i][j];

        float Ro[3][3], to[3];
        for (int i = 0; i < 9; ++i) Ro[i / 3][i % 3] = pose[i];
        for (int i = 0; i < 3; ++i) to[i] = pose[9 + i];

        float Rn[3][3], tn[3];
        for (int i = 0; i < 3; ++i) {
            for (int j = 0; j < 3; ++j) {
                float s = 0.f;
                for (int k = 0; k < 3; ++k) s += Rd[i][k] * Ro[k][j];
                Rn[i][j] = s;
            }
            float s = 0.f;
            for (int k = 0; k < 3; ++k) s += Rd[i][k] * to[k];
            tn[i] = s + delta[3 + i];
        }
        for (int i = 0; i < 9; ++i) { pose[i] = Rn[i / 3][i % 3]; outp[i] = Rn[i / 3][i % 3]; }
        for (int i = 0; i < 3; ++i) { pose[9 + i] = tn[i]; outp[9 + i] = tn[i]; }
    }
}

extern "C" void kernel_launch(void* const* d_in, const int* in_sizes, int n_in,
                              void* d_out, int out_size, void* d_ws, size_t ws_size,
                              hipStream_t stream) {
    const float* pts3D = (const float*)d_in[0];
    const float* fref  = (const float*)d_in[1];
    const float* fq    = (const float*)d_in[2];
    const float* fgx   = (const float*)d_in[3];
    const float* fgy   = (const float*)d_in[4];
    const float* Km    = (const float*)d_in[5];
    const float* Rinit = (const float*)d_in[6];
    const float* tinit = (const float*)d_in[7];
    const int*   imwP  = (const int*)d_in[8];
    const int*   imhP  = (const int*)d_in[9];

    const int nPts    = in_sizes[0] / 3;
    const int nTiles  = (nPts + 15) / 16;
    const int nBlocks = (nTiles + 7) / 8;

    float* pose     = (float*)d_ws;      // 12 floats (reserve 64)
    float* partials = pose + 64;         // nBlocks*27 floats

    // Channel-contiguous scratch copy of the three maps, if workspace allows.
    const size_t T_ELEMS = (size_t)3 * CC * HW;
    size_t tOff = ((size_t)64 + (size_t)nBlocks * 27 + 63) & ~(size_t)63;
    const bool fast = ((tOff + T_ELEMS) * sizeof(float)) <= ws_size;
    float* T = pose + tOff;

    pnp_init_pose<<<1, 32, 0, stream>>>(Rinit, tinit, pose);
    if (fast) {
        const int tpBlocks = (CC / 32) * (WW / 32) * HH;   // 28800
        pnp_transpose<<<tpBlocks, 256, 0, stream>>>(fq,  T, 0);
        pnp_transpose<<<tpBlocks, 256, 0, stream>>>(fgx, T, 1);
        pnp_transpose<<<tpBlocks, 256, 0, stream>>>(fgy, T, 2);
    }
    for (int it = 0; it < N_ITERS; ++it) {
        if (fast)
            pnp_accum<true><<<nBlocks, 256, 0, stream>>>(pts3D, fref, fq, fgx, fgy, T, Km, pose,
                                                         partials, imwP, imhP, nPts);
        else
            pnp_accum<false><<<nBlocks, 256, 0, stream>>>(pts3D, fref, fq, fgx, fgy, T, Km, pose,
                                                          partials, imwP, imhP, nPts);
        pnp_solve_update<<<1, 32, 0, stream>>>(partials, nBlocks, pose, (float*)d_out);
    }
}
